// Block_38388417692234
// MI455X (gfx1250) — matrix operand order repaired and checked by reading, not compiled, not run on hardware
//
#include <hip/hip_runtime.h>
#include <math.h>

// ---------------------------------------------------------------------------
// ViT block forward for MI455X (gfx1250): all GEMMs via v_wmma_f32_16x16x32_bf16
// ---------------------------------------------------------------------------

typedef __attribute__((ext_vector_type(16))) __bf16 v16bf;
typedef __attribute__((ext_vector_type(8)))  __bf16 v8bf;
typedef __attribute__((ext_vector_type(8)))  float  v8f;

constexpr int Bt    = 32;
constexpr int Nn    = 512;
constexpr int Cc    = 768;
constexpr int Hh    = 12;
constexpr int Dh    = 64;
constexpr int HIDn  = 3072;
constexpr int Mrows = Bt * Nn;           // 16384 token rows

#define DEVFN static __device__ __forceinline__

// float -> bf16 bits via the native converter (single v_cvt op on gfx1250)
DEVFN unsigned short f2bf(float f) {
  return __builtin_bit_cast(unsigned short, (__bf16)f);
}

// pack two floats into two bf16 lanes of one dword
DEVFN unsigned f2bf2(float lo, float hi) {
  return (unsigned)f2bf(lo) | ((unsigned)f2bf(hi) << 16);
}

DEVFN float bf2f(unsigned short b) {
  union { unsigned u; float f; } v; v.u = (unsigned)b << 16; return v.f;
}

// Build the 16x32 A-operand (contiguous K order, matching B):
// lane l (l<16):  row = l, K = {0..15}
// lane l (l>=16): row = l-16, K = {16..31}
DEVFN v16bf make_a(const unsigned short* p0, const unsigned short* p1) {
  v8bf a0 = *(const v8bf*)p0;
  v8bf a1 = *(const v8bf*)p1;
  return __builtin_shufflevector(a0, a1, 0,1,2,3,4,5,6,7,8,9,10,11,12,13,14,15);
}

DEVFN v8f wmma_bf16(v16bf a, v16bf b, v8f c) {
  return __builtin_amdgcn_wmma_f32_16x16x32_bf16(
      /*neg_a=*/false, a, /*neg_b=*/false, b,
      /*c_mod=*/(short)0, c, /*reuse_a=*/false, /*reuse_b=*/false);
}

// ---------------------------------------------------------------------------
// f32 -> bf16 weight conversion, 2 elements per thread, packed dword store
// (all weight tensor sizes are even)
// ---------------------------------------------------------------------------
__global__ void cvt_bf16_kernel(const float* __restrict__ src,
                                unsigned* __restrict__ dst, int npairs) {
  int i = blockIdx.x * blockDim.x + threadIdx.x;
  if (i < npairs) {
    float2 v = ((const float2*)src)[i];
    dst[i] = f2bf2(v.x, v.y);
  }
}

// ---------------------------------------------------------------------------
// LayerNorm over C=768, one wave (32 lanes) per row; lane owns 12 channel
// pairs -> b64 loads, packed b32 bf16 stores.
// ---------------------------------------------------------------------------
__global__ __launch_bounds__(256)
void ln_kernel(const float* __restrict__ x, const float* __restrict__ w,
               const float* __restrict__ b, unsigned short* __restrict__ out) {
  const int wave = threadIdx.x >> 5;
  const int lane = threadIdx.x & 31;
  const int row  = blockIdx.x * 8 + wave;
  const float2* xr = (const float2*)(x + (size_t)row * Cc);

  float2 vals[12];
  float s = 0.f, ss = 0.f;
#pragma unroll
  for (int i = 0; i < 12; ++i) {
    float2 v = xr[lane + 32 * i];
    vals[i] = v;
    s += v.x + v.y; ss += v.x * v.x + v.y * v.y;
  }
#pragma unroll
  for (int off = 16; off > 0; off >>= 1) {
    s  += __shfl_xor(s,  off, 32);
    ss += __shfl_xor(ss, off, 32);
  }
  const float mu  = s * (1.f / Cc);
  const float var = ss * (1.f / Cc) - mu * mu;
  const float rs  = rsqrtf(var + 1e-5f);
  unsigned* orow = (unsigned*)(out + (size_t)row * Cc);
#pragma unroll
  for (int i = 0; i < 12; ++i) {
    const int cp = lane + 32 * i;          // channel pair index
    const float a0 = (vals[i].x - mu) * rs * w[2 * cp]     + b[2 * cp];
    const float a1 = (vals[i].y - mu) * rs * w[2 * cp + 1] + b[2 * cp + 1];
    orow[cp] = f2bf2(a0, a1);
  }
}

// ---------------------------------------------------------------------------
// Generic bf16 GEMM: out[m][n] = sum_k A[m][k] * W[n][k]  (+ epilogue)
// 128 threads = 4 waves; each wave owns a 32(M) x 64(N) register tile
// (2 A tiles, 4 B tiles, 8 WMMAs per 32-wide K chunk). WG tile 32 x 256.
// MODE 0: qkv scatter (Q pre-scaled by D^-0.5; Q,K [BH,N,D]; V^T [BH,D,N]) bf16
// MODE 1: proj: x2 = 2*(acc + bias) -> f32
// MODE 2: fc1 : gelu(acc + bias)    -> bf16
// MODE 3: fc2 : resid + acc + bias  -> f32 (final output)
// ---------------------------------------------------------------------------
template <int MODE>
__global__ __launch_bounds__(128)
void gemm16(const unsigned short* __restrict__ A,
            const unsigned short* __restrict__ W,
            const float* __restrict__ bias,
            float* __restrict__ outf,
            unsigned short* __restrict__ outbf,
            unsigned short* __restrict__ qb,
            unsigned short* __restrict__ kb,
            unsigned short* __restrict__ vb,
            const float* __restrict__ resid,
            int K) {
  const int lane = threadIdx.x & 31;
  const int wave = threadIdx.x >> 5;
  const int half = lane >> 4;
  const int l15  = lane & 15;
  const int m0   = blockIdx.y * 32;
  const int n0   = blockIdx.x * 256 + wave * 64;

  const unsigned short* arow0 = A + (size_t)(m0 + l15) * K;
  const unsigned short* arow1 = A + (size_t)(m0 + 16 + l15) * K;

  v8f acc[2][4] = {};
  for (int k0 = 0; k0 < K; k0 += 32) {
    v16bf A0 = make_a(arow0 + k0 + half * 16, arow0 + k0 + half * 16 + 8);
    v16bf A1 = make_a(arow1 + k0 + half * 16, arow1 + k0 + half * 16 + 8);
    if (k0 + 256 < K) {                     // stream-ahead hint (global_prefetch)
      __builtin_prefetch(arow0 + k0 + 256, 0, 1);
      __builtin_prefetch(arow1 + k0 + 256, 0, 1);
    }
#pragma unroll
    for (int t = 0; t < 4; ++t) {
      const int n = n0 + t * 16 + l15;
      v16bf Bi = *(const v16bf*)(W + (size_t)n * K + k0 + half * 16);
      acc[0][t] = wmma_bf16(A0, Bi, acc[0][t]);
      acc[1][t] = wmma_bf16(A1, Bi, acc[1][t]);
    }
  }

  // Epilogue: C/D layout -> element (M = r + half*8, N = l15) of each tile
#pragma unroll
  for (int mt = 0; mt < 2; ++mt) {
#pragma unroll
    for (int t = 0; t < 4; ++t) {
      const int col = n0 + t * 16 + l15;
#pragma unroll
      for (int r = 0; r < 8; ++r) {
        const int row = m0 + mt * 16 + r + half * 8;
        const float a = acc[mt][t][r];
        if constexpr (MODE == 0) {
          const int part = col / Cc;         // 0=q 1=k 2=v
          const int rem  = col % Cc;
          const int h    = rem / Dh;
          const int d    = rem % Dh;
          const int bb   = row >> 9;         // row / N  (N = 512)
          const int nq   = row & (Nn - 1);   // row % N
          const size_t bh = (size_t)bb * Hh + h;
          if (part == 0)      qb[(bh * Nn + nq) * Dh + d] = f2bf(a * 0.125f);
          else if (part == 1) kb[(bh * Nn + nq) * Dh + d] = f2bf(a);
          else                vb[(bh * Dh + d) * Nn + nq] = f2bf(a); // V^T
        } else if constexpr (MODE == 1) {
          outf[(size_t)row * Cc + col] = 2.f * (a + bias[col]);
        } else if constexpr (MODE == 2) {
          const float v = a + bias[col];
          const float g = 0.5f * v * (1.f + erff(v * 0.70710678118654752f));
          outbf[(size_t)row * HIDn + col] = f2bf(g);
        } else {
          outf[(size_t)row * Cc + col] =
              resid[(size_t)row * Cc + col] + a + bias[col];
        }
      }
    }
  }
}

// ---------------------------------------------------------------------------
// Attention: one wave per (b, h, 16-query tile).
// Single in-place bf16 S/P tile (16 x 512, rows padded to 520) in LDS:
// phase 1 writes logits (Q pre-scaled), phase 2 overwrites with exp(S - max),
// phase 3 consumes it as the WMMA A operand. ~17 KB LDS per wave.
// ---------------------------------------------------------------------------
__global__ __launch_bounds__(32)
void attn_kernel(const unsigned short* __restrict__ q,
                 const unsigned short* __restrict__ k,
                 const unsigned short* __restrict__ vt,
                 unsigned short* __restrict__ o) {
  __shared__ unsigned short SP[16][520];   // padded: 16B rows, spread banks
  __shared__ float          red[2][16];
  __shared__ float          invs[16];

  const int lane = threadIdx.x;
  const int half = lane >> 4;
  const int l15  = lane & 15;
  const int qt = blockIdx.x & 31;          // query tile (N/16 = 32)
  const int bh = blockIdx.x >> 5;
  const int bb = bh / Hh;
  const int hh = bh % Hh;

  const unsigned short* Q = q  + (size_t)bh * Nn * Dh;
  const unsigned short* K = k  + (size_t)bh * Nn * Dh;
  const unsigned short* V = vt + (size_t)bh * Dh * Nn;  // [D][N]

  // ---- Phase 1: S = (scale*Q) Kt ----
  const unsigned short* qrow = Q + (size_t)(qt * 16 + l15) * Dh;
  v16bf Aq[2];
#pragma unroll
  for (int kc = 0; kc < 2; ++kc)
    Aq[kc] = make_a(qrow + kc * 32 + half * 16, qrow + kc * 32 + half * 16 + 8);

  for (int kt = 0; kt < 32; ++kt) {
    v8f acc = {};
#pragma unroll
    for (int kc = 0; kc < 2; ++kc) {
      // B = Kt tile: column n = key (kt*16+l15), K-dim = d (contiguous in K rows)
      v16bf Bi = *(const v16bf*)(K + (size_t)(kt * 16 + l15) * Dh +
                                 kc * 32 + half * 16);
      acc = wmma_bf16(Aq[kc], Bi, acc);
    }
#pragma unroll
    for (int r = 0; r < 8; ++r)
      SP[r + half * 8][kt * 16 + l15] = f2bf(acc[r]);
  }
  __syncthreads();

  // ---- Phase 2: row softmax, in place (bf16 logits -> bf16 probabilities) ----
  const int row = l15, seg = half;
  float mx = -1e30f;
  for (int i = 0; i < 256; ++i) mx = fmaxf(mx, bf2f(SP[row][seg * 256 + i]));
  red[seg][row] = mx;
  __syncthreads();
  const float m = fmaxf(red[0][row], red[1][row]);
  __syncthreads();

  float sum = 0.f;
  const float l2e = 1.44269504088896f;
  for (int i = 0; i < 256; ++i) {
    float p = exp2f((bf2f(SP[row][seg * 256 + i]) - m) * l2e);
    sum += p;
    SP[row][seg * 256 + i] = f2bf(p);
  }
  red[seg][row] = sum;
  __syncthreads();
  if (half == 0) invs[row] = 1.f / (red[0][row] + red[1][row]);
  __syncthreads();

  // ---- Phase 3: O = (P / sum) V ----
  v8f oacc[4] = {};
  for (int kc2 = 0; kc2 < 16; ++kc2) {     // 512 keys in chunks of 32
    const unsigned short* prow = &SP[l15][kc2 * 32];
    v16bf Ai = make_a(prow + half * 16, prow + half * 16 + 8);
#pragma unroll
    for (int t = 0; t < 4; ++t) {
      // B = V chunk: column n = d (t*16+l15) -> row of V^T, keys contiguous
      v16bf Bi = *(const v16bf*)(V + (size_t)(t * 16 + l15) * Nn +
                                 kc2 * 32 + half * 16);
      oacc[t] = wmma_bf16(Ai, Bi, oacc[t]);
    }
  }

  // o[b, nq, h*D + d] row-major [M, C], bf16
#pragma unroll
  for (int t = 0; t < 4; ++t) {
#pragma unroll
    for (int r = 0; r < 8; ++r) {
      const int mq = qt * 16 + r + half * 8;
      const int d  = t * 16 + l15;
      const float val = oacc[t][r] * invs[r + half * 8];
      o[((size_t)(bb * Nn + mq)) * Cc + hh * Dh + d] = f2bf(val);
    }
  }
}

// ---------------------------------------------------------------------------
// Workspace layout (bytes)
// ---------------------------------------------------------------------------
constexpr size_t al256(size_t x) { return (x + 255) & ~(size_t)255; }
constexpr size_t SZ_WQKV = (size_t)3 * Cc * Cc * 2;
constexpr size_t SZ_WPRJ = (size_t)Cc * Cc * 2;
constexpr size_t SZ_WFC1 = (size_t)HIDn * Cc * 2;
constexpr size_t SZ_WFC2 = (size_t)Cc * HIDn * 2;
constexpr size_t SZ_ACT  = (size_t)Mrows * Cc * 2;     // bf16 [M, C]
constexpr size_t SZ_X2   = (size_t)Mrows * Cc * 4;     // f32  [M, C]
constexpr size_t SZ_HID  = (size_t)Mrows * HIDn * 2;   // bf16 [M, HID]

constexpr size_t OFF_WQKV = 0;
constexpr size_t OFF_WPRJ = OFF_WQKV + al256(SZ_WQKV);
constexpr size_t OFF_WFC1 = OFF_WPRJ + al256(SZ_WPRJ);
constexpr size_t OFF_WFC2 = OFF_WFC1 + al256(SZ_WFC1);
constexpr size_t OFF_H1   = OFF_WFC2 + al256(SZ_WFC2); // LN1 out; reused as attn O
constexpr size_t OFF_Q    = OFF_H1   + al256(SZ_ACT);  // reused as LN2 out
constexpr size_t OFF_K    = OFF_Q    + al256(SZ_ACT);
constexpr size_t OFF_VT   = OFF_K    + al256(SZ_ACT);
constexpr size_t OFF_X2   = OFF_VT   + al256(SZ_ACT);
constexpr size_t OFF_HID  = OFF_X2   + al256(SZ_X2);

extern "C" void kernel_launch(void* const* d_in, const int* in_sizes, int n_in,
                              void* d_out, int out_size, void* d_ws, size_t ws_size,
                              hipStream_t stream) {
  const float* x      = (const float*)d_in[0];
  const float* ln1_w  = (const float*)d_in[1];
  const float* ln1_b  = (const float*)d_in[2];
  const float* qkv_w  = (const float*)d_in[3];
  const float* proj_w = (const float*)d_in[4];
  const float* proj_b = (const float*)d_in[5];
  const float* ln2_w  = (const float*)d_in[6];
  const float* ln2_b  = (const float*)d_in[7];
  const float* fc1_w  = (const float*)d_in[8];
  const float* fc1_b  = (const float*)d_in[9];
  const float* fc2_w  = (const float*)d_in[10];
  const float* fc2_b  = (const float*)d_in[11];
  float* out = (float*)d_out;

  char* ws = (char*)d_ws;
  unsigned short* wqkv = (unsigned short*)(ws + OFF_WQKV);
  unsigned short* wprj = (unsigned short*)(ws + OFF_WPRJ);
  unsigned short* wfc1 = (unsigned short*)(ws + OFF_WFC1);
  unsigned short* wfc2 = (unsigned short*)(ws + OFF_WFC2);
  unsigned short* h1   = (unsigned short*)(ws + OFF_H1);
  unsigned short* ob   = h1;                              // reuse (h1 dead post-qkv)
  unsigned short* qb   = (unsigned short*)(ws + OFF_Q);
  unsigned short* h2a  = qb;                              // reuse (q dead post-attn)
  unsigned short* kb   = (unsigned short*)(ws + OFF_K);
  unsigned short* vb   = (unsigned short*)(ws + OFF_VT);
  float*          x2   = (float*)(ws + OFF_X2);
  unsigned short* hid  = (unsigned short*)(ws + OFF_HID);

  // 1) weights -> bf16 (paired)
  auto cvt = [&](const float* s, unsigned short* d, int n) {
    int np = n / 2;
    cvt_bf16_kernel<<<(np + 255) / 256, 256, 0, stream>>>(s, (unsigned*)d, np);
  };
  cvt(qkv_w,  wqkv, 3 * Cc * Cc);
  cvt(proj_w, wprj, Cc * Cc);
  cvt(fc1_w,  wfc1, HIDn * Cc);
  cvt(fc2_w,  wfc2, Cc * HIDn);

  // 2) LN1
  ln_kernel<<<Mrows / 8, 256, 0, stream>>>(x, ln1_w, ln1_b, h1);

  // 3) QKV GEMM (Nout = 2304)
  gemm16<0><<<dim3(3 * Cc / 256, Mrows / 32), 128, 0, stream>>>(
      h1, wqkv, nullptr, nullptr, nullptr, qb, kb, vb, nullptr, Cc);

  // 4) attention
  attn_kernel<<<Bt * Hh * (Nn / 16), 32, 0, stream>>>(qb, kb, vb, ob);

  // 5) proj GEMM -> x2 = 2*(oW^T + b)
  gemm16<1><<<dim3(Cc / 256, Mrows / 32), 128, 0, stream>>>(
      ob, wprj, proj_b, x2, nullptr, nullptr, nullptr, nullptr, nullptr, Cc);

  // 6) LN2
  ln_kernel<<<Mrows / 8, 256, 0, stream>>>(x2, ln2_w, ln2_b, h2a);

  // 7) FC1 + GELU (Nout = 3072)
  gemm16<2><<<dim3(HIDn / 256, Mrows / 32), 128, 0, stream>>>(
      h2a, wfc1, fc1_b, nullptr, hid, nullptr, nullptr, nullptr, nullptr, Cc);

  // 8) FC2 + bias + residual -> out (K = 3072)
  gemm16<3><<<dim3(Cc / 256, Mrows / 32), 128, 0, stream>>>(
      hid, wfc2, fc2_b, out, nullptr, nullptr, nullptr, nullptr, x2, HIDn);
}
